// SelfAttention_50465865728509
// MI455X (gfx1250) — matrix operand order repaired and checked by reading, not compiled, not run on hardware
//
#include <hip/hip_runtime.h>
#include <hip/hip_bf16.h>

typedef __attribute__((ext_vector_type(16))) _Float16 v16h;
typedef __attribute__((ext_vector_type(8)))  float    v8f;
typedef unsigned int v4u __attribute__((ext_vector_type(4)));
typedef int          v8i __attribute__((ext_vector_type(8)));
typedef int          v4i __attribute__((ext_vector_type(4)));

#define NB_ 8
#define T_  2048
#define D_  1024
#define DK_ 1024
#define DV_ 1024

#if __has_builtin(__builtin_amdgcn_tensor_load_to_lds)
#define HAVE_TDM 1
#else
#define HAVE_TDM 0
#endif

// ---------------------------------------------------------------- f32 -> f16
__global__ __launch_bounds__(256)
void cvt_f32_to_f16(const float* __restrict__ s, _Float16* __restrict__ d, long long n) {
    long long i = ((long long)blockIdx.x * 256 + threadIdx.x) * 4;
    if (i + 3 < n) {
        float4 v = *(const float4*)(s + i);
        union { _Float16 h[4]; uint2 u; } p;
        p.h[0] = (_Float16)v.x; p.h[1] = (_Float16)v.y;
        p.h[2] = (_Float16)v.z; p.h[3] = (_Float16)v.w;
        *(uint2*)(d + i) = p.u;
    }
}

#if HAVE_TDM
// TDM: DMA a [rows x 32] f16 tile (row stride ldElems) into LDS at ldsByteOff,
// padding 12 dwords after every 16 dwords -> 56-half LDS pitch (bank-conflict free).
// D# built per CDNA5 ISA 8.3/8.4 (data_size=2B, type=2, count=1, 2D tile).
// Toolchain uses the 6-arg builtin: (v4u, v8i, v4i, v4i, v8i, i32 cpol).
__device__ __forceinline__ void tdm_load_tile_2d(unsigned ldsByteOff,
                                                 const _Float16* gsrc,
                                                 unsigned rows, unsigned ldElems)
{
    unsigned long long ga = (unsigned long long)(size_t)(const void*)gsrc;
    v4u g0;
    g0[0] = 1u;                                                  // count=1 (valid)
    g0[1] = ldsByteOff;                                          // lds_addr
    g0[2] = (unsigned)ga;                                        // global_addr[31:0]
    g0[3] = (unsigned)((ga >> 32) & 0x1ffffffull) | (2u << 30);  // ga[56:32] | type=2
    v8i g1;
    g1[0] = (int)((1u << 16)       // data_size = 2 bytes
                | (1u << 20)       // pad_enable
                | (3u << 22)       // pad_interval: every 16 dwords (one 32-f16 row)
                | (11u << 25));    // pad_amount: 12 dwords (48B) -> pitch 56 halves
    g1[1] = (int)(32u << 16);                          // tensor_dim0 lo = 32 (k extent)
    g1[2] = (int)((rows & 0xffffu) << 16);             // dim0 hi = 0 | tensor_dim1 lo = rows
    g1[3] = (int)(32u << 16);                          // dim1 hi = 0 | tile_dim0 = 32
    g1[4] = (int)(rows & 0xffffu);                     // tile_dim1 = rows, tile_dim2 = 0
    g1[5] = (int)ldElems;                              // tensor_dim0_stride[31:0]
    g1[6] = 0;                                         // stride0 hi | stride1 lo
    g1[7] = 0;                                         // stride1 hi
    v4i z4 = {0, 0, 0, 0};                             // groups 2/3 unused (2D tensor)
    v8i z8 = {0, 0, 0, 0, 0, 0, 0, 0};
    __builtin_amdgcn_tensor_load_to_lds(g0, g1, z4, z4, z8, 0);
}
#endif

// ---------------------------------------------------------------- WMMA GEMM
// C[m,n] = sum_k A[m,k] * B(k,n)   (optionally relu(. + bias[n]))
// B_NT=false: B stored [k][n] (manual LDS-transpose staging)
// B_NT=true : B stored [n][k] => computes A*B^T (TDM-staged like A)
// Block: 256 thr = 8 waves. Tile M=128, N=128, Kstep=32, double-buffered LDS.
// Wave (wm=wave&3, wn=wave>>2) owns a 32x64 sub-tile: 8 WMMAs per K-step.
template<bool B_NT, bool BIAS_RELU, bool OUT_F16>
__global__ __launch_bounds__(256)
void gemm_wmma_k(const _Float16* __restrict__ A, long long aBatch, int lda,
                 const _Float16* __restrict__ B, long long bBatch, int ldb,
                 void* __restrict__ Cptr, long long cBatch, int ldc,
                 int Kdim,
                 const float* __restrict__ b0,
                 const float* __restrict__ b1,
                 const float* __restrict__ b2)
{
    constexpr int PITCH = 56;               // 112B rows: 16B aligned, conflict-free
    constexpr int TILEH = 128 * PITCH;      // halves per tile buffer
    __shared__ __align__(16) _Float16 smem[4 * TILEH];   // A0 A1 B0 B1

    const int tid  = threadIdx.x;
    const int lane = tid & 31;
    const int wave = tid >> 5;
    const int wm   = wave & 3;              // 4 M-slots of 32 rows
    const int wn   = wave >> 2;             // 2 N-slots of 64 cols
    const int z    = blockIdx.z;
    const int nblk = blockIdx.x * 128;

    const _Float16* Ab = A + (long long)z * aBatch + (long long)blockIdx.y * 128 * lda;
    const _Float16* Bb = B + (long long)z * bBatch;
    if constexpr (B_NT) Bb += (long long)nblk * ldb;

    v8f acc[2][4] = {};
    const int ntiles = Kdim >> 5;

#if HAVE_TDM
    const unsigned ldsBase = (unsigned)(size_t)(const void*)&smem[0];
    if (wave == 0) {        // prologue: DMA tile 0 into buffer 0
        tdm_load_tile_2d(ldsBase, Ab, 128u, (unsigned)lda);
        if constexpr (B_NT)
            tdm_load_tile_2d(ldsBase + 2u * TILEH * (unsigned)sizeof(_Float16),
                             Bb, 128u, (unsigned)ldb);
    }
#endif

    for (int it = 0; it < ntiles; ++it) {
        const int k0  = it << 5;
        const int buf = it & 1;
        _Float16* As = smem + buf * TILEH;
        _Float16* Bs = smem + (2 + buf) * TILEH;

#if HAVE_TDM
        if (wave == 0) {
            __builtin_amdgcn_s_wait_tensorcnt(0);           // tile `it` landed in LDS
            if (it + 1 < ntiles) {                          // prefetch tile it+1
                const int nb = buf ^ 1;
                tdm_load_tile_2d(ldsBase + (unsigned)(nb * TILEH) * (unsigned)sizeof(_Float16),
                                 Ab + (k0 + 32), 128u, (unsigned)lda);
                if constexpr (B_NT)
                    tdm_load_tile_2d(ldsBase + (unsigned)((2 + nb) * TILEH) * (unsigned)sizeof(_Float16),
                                     Bb + (k0 + 32), 128u, (unsigned)ldb);
            }
        }
#else
        {   // manual A stage: 128x32, two 16B chunks per thread
            int e0 = tid * 8;
            int r0 = e0 >> 5, c0 = e0 & 31;
            *(uint4*)&As[r0 * PITCH + c0] = *(const uint4*)&Ab[(long long)r0 * lda + k0 + c0];
            int e1 = e0 + 2048;
            int r1 = e1 >> 5, c1 = e1 & 31;
            *(uint4*)&As[r1 * PITCH + c1] = *(const uint4*)&Ab[(long long)r1 * lda + k0 + c1];
        }
        if constexpr (B_NT) {   // B already [n][k]: 128x32 straight copy
            int n = tid >> 1, c = (tid & 1) * 16;
            *(uint4*)&Bs[n * PITCH + c]     = *(const uint4*)&Bb[(long long)n * ldb + k0 + c];
            *(uint4*)&Bs[n * PITCH + c + 8] = *(const uint4*)&Bb[(long long)n * ldb + k0 + c + 8];
        }
#endif
        if constexpr (!B_NT) {  // B is [k][n]: transpose into Bs[n][k] (32x128 tile)
            int kk = tid >> 3, nb0 = (tid & 7) * 16;
            union { uint4 q; _Float16 h[8]; } u0, u1;
            u0.q = *(const uint4*)&Bb[(long long)(k0 + kk) * ldb + nblk + nb0];
            u1.q = *(const uint4*)&Bb[(long long)(k0 + kk) * ldb + nblk + nb0 + 8];
            #pragma unroll
            for (int j = 0; j < 8; ++j) {
                Bs[(nb0 + j) * PITCH + kk]     = u0.h[j];
                Bs[(nb0 + j + 8) * PITCH + kk] = u1.h[j];
            }
        }
        __syncthreads();

        // ---- load all fragments first, then a back-to-back WMMA burst
        union F { v16h v; uint4 q[2]; };
        const int hv = lane >> 4;
        const int ml = lane & 15;
        F af[2], bf[4];
        #pragma unroll
        for (int mt = 0; mt < 2; ++mt) {
            const int ar = wm * 32 + mt * 16 + ml;
            af[mt].q[0] = *(const uint4*)&As[ar * PITCH + hv * 8];
            af[mt].q[1] = *(const uint4*)&As[ar * PITCH + 16 + hv * 8];
        }
        #pragma unroll
        for (int nt = 0; nt < 4; ++nt) {
            const int bn = wn * 64 + nt * 16 + ml;
            bf[nt].q[0] = *(const uint4*)&Bs[bn * PITCH + hv * 8];
            bf[nt].q[1] = *(const uint4*)&Bs[bn * PITCH + 16 + hv * 8];
        }
        #pragma unroll
        for (int mt = 0; mt < 2; ++mt)
            #pragma unroll
            for (int nt = 0; nt < 4; ++nt)
                acc[mt][nt] = __builtin_amdgcn_wmma_f32_16x16x32_f16(
                    false, af[mt].v, false, bf[nt].v, (short)0, acc[mt][nt], false, false);
        __syncthreads();
    }

    // ---- epilogue: C layout VGPR i -> M = i + 8*(lane>=16), N = lane&15
    const float* bias = (z == 0) ? b0 : (z == 1) ? b1 : b2;
    const int nl = lane & 15;
    #pragma unroll
    for (int mt = 0; mt < 2; ++mt) {
        const long long mbase = (long long)blockIdx.y * 128 + wm * 32 + mt * 16 + (lane >> 4) * 8;
        #pragma unroll
        for (int nt = 0; nt < 4; ++nt) {
            const int ncol = nblk + wn * 64 + nt * 16 + nl;
            float badd = 0.f;
            if constexpr (BIAS_RELU) badd = bias[ncol];
            #pragma unroll
            for (int i = 0; i < 8; ++i) {
                float v = acc[mt][nt][i];
                if constexpr (BIAS_RELU) v = fmaxf(v + badd, 0.f);
                const long long cidx = (long long)z * cBatch + (mbase + i) * ldc + ncol;
                if constexpr (OUT_F16) ((_Float16*)Cptr)[cidx] = (_Float16)v;
                else                   ((float*)Cptr)[cidx]    = v;
            }
        }
    }
}

// ---------------------------------------------------------------- row softmax
__global__ __launch_bounds__(256)
void softmax_rows(_Float16* __restrict__ S) {
    _Float16* p = S + (long long)blockIdx.x * T_;
    const int tid = threadIdx.x, lane = tid & 31, wave = tid >> 5;
    __shared__ float red[8];

    union { uint4 q; _Float16 h[8]; } u;
    u.q = *(const uint4*)&p[tid * 8];
    float f[8];
    #pragma unroll
    for (int i = 0; i < 8; ++i) f[i] = (float)u.h[i];

    float m = f[0];
    #pragma unroll
    for (int i = 1; i < 8; ++i) m = fmaxf(m, f[i]);
    #pragma unroll
    for (int off = 16; off > 0; off >>= 1) m = fmaxf(m, __shfl_xor(m, off, 32));
    if (lane == 0) red[wave] = m;
    __syncthreads();
    if (wave == 0) {
        float t = red[lane & 7];
        #pragma unroll
        for (int off = 4; off > 0; off >>= 1) t = fmaxf(t, __shfl_xor(t, off, 32));
        if (lane == 0) red[0] = t;
    }
    __syncthreads();
    m = red[0];
    __syncthreads();

    float s = 0.f;
    #pragma unroll
    for (int i = 0; i < 8; ++i) { f[i] = __expf(f[i] - m); s += f[i]; }
    #pragma unroll
    for (int off = 16; off > 0; off >>= 1) s += __shfl_xor(s, off, 32);
    if (lane == 0) red[wave] = s;
    __syncthreads();
    if (wave == 0) {
        float t = red[lane & 7];
        #pragma unroll
        for (int off = 4; off > 0; off >>= 1) t += __shfl_xor(t, off, 32);
        if (lane == 0) red[0] = t;
    }
    __syncthreads();
    const float inv = 1.0f / red[0];
    #pragma unroll
    for (int i = 0; i < 8; ++i) u.h[i] = (_Float16)(f[i] * inv);
    *(uint4*)&p[tid * 8] = u.q;
}

// ---------------------------------------------------------------- launch
extern "C" void kernel_launch(void* const* d_in, const int* in_sizes, int n_in,
                              void* d_out, int out_size, void* d_ws, size_t ws_size,
                              hipStream_t stream) {
    const float* X  = (const float*)d_in[0];
    const float* Wq = (const float*)d_in[1];
    const float* bq = (const float*)d_in[2];
    const float* Wk = (const float*)d_in[3];
    const float* bk = (const float*)d_in[4];
    const float* Wv = (const float*)d_in[5];
    const float* bv = (const float*)d_in[6];

    const long long M = (long long)NB_ * T_;          // 16384 rows
    // workspace (f16): Xh 32MB | Wh 6MB | Q 32MB | K 32MB | V 32MB | S 64MB
    _Float16* Xh = (_Float16*)d_ws;
    _Float16* Wh = Xh + M * D_;
    _Float16* Qh = Wh + 3ll * D_ * DK_;
    _Float16* Kh = Qh + M * DK_;
    _Float16* Vh = Kh + M * DK_;
    _Float16* Sh = Vh + M * DV_;

    // 0) casts to f16
    {
        long long n = M * D_;
        cvt_f32_to_f16<<<dim3((unsigned)(n / 1024)), 256, 0, stream>>>(X, Xh, n);
        long long wn = (long long)D_ * DK_;
        cvt_f32_to_f16<<<dim3((unsigned)(wn / 1024)), 256, 0, stream>>>(Wq, Wh,          wn);
        cvt_f32_to_f16<<<dim3((unsigned)(wn / 1024)), 256, 0, stream>>>(Wk, Wh +     wn, wn);
        cvt_f32_to_f16<<<dim3((unsigned)(wn / 1024)), 256, 0, stream>>>(Wv, Wh + 2 * wn, wn);
    }

    // 1) Q/K/V = relu(X @ W + b)   (z selects q/k/v)
    gemm_wmma_k<false, true, true><<<dim3(DK_ / 128, (unsigned)(M / 128), 3), 256, 0, stream>>>(
        Xh, 0ll, D_,
        Wh, (long long)D_ * DK_, DK_,
        Qh, M * (long long)DK_, DK_,
        D_, bq, bk, bv);

    // 2) per batch: S = Q @ K^T  (B stored [n][k] -> TDM-staged transposed path)
    gemm_wmma_k<true, false, true><<<dim3(T_ / 128, T_ / 128, NB_), 256, 0, stream>>>(
        Qh, (long long)T_ * DK_, DK_,
        Kh, (long long)T_ * DK_, DK_,
        Sh, (long long)T_ * T_, T_,
        DK_, nullptr, nullptr, nullptr);

    // 3) softmax over each 2048-wide score row, in place
    softmax_rows<<<dim3(NB_ * T_), 256, 0, stream>>>(Sh);

    // 4) per batch: out = P @ V  (K-dim = T = 2048), f32 out
    gemm_wmma_k<false, false, false><<<dim3(DV_ / 128, T_ / 128, NB_), 256, 0, stream>>>(
        Sh, (long long)T_ * T_, T_,
        Vh, (long long)T_ * DV_, DV_,
        d_out, (long long)T_ * DV_, DV_,
        T_, nullptr, nullptr, nullptr);
}
